// MHAttention_72086731096520
// MI455X (gfx1250) — matrix operand order repaired and checked by reading, not compiled, not run on hardware
//
#include <hip/hip_runtime.h>
#include <hip/hip_bf16.h>
#include <math.h>

// ---------------------------------------------------------------------------
// Linformer MHA for MI455X (gfx1250, wave32, WMMA bf16 16x16x32, f32 accum)
// B=8 S=4096 C=1024 H=16 D=64 K=256
//
// Pipeline:
//   1. fuse to_{q,k,v}_w with per-head w{q,k,v}_w  -> 3x [1024,1024] bf16
//   2. three NT WMMA GEMMs: tensor x Wf -> QW [B,H,S,D], KWt/VWt [B,H,D,S]
//   3. two NT WMMA GEMMs (reduce S=4096): KEt [B,H,K,D], VFt [B,H,D,K]
//   4. fused attention: per (b,h,S-tile) stage KEt/VFt in LDS via TDM,
//      WMMA scores, LDS softmax, WMMA P*V from LDS, GELU, concat scatter
//   5. NT WMMA GEMM: gelu(concat) x wo_w^T + wo_b -> f32 out
// ---------------------------------------------------------------------------

typedef __attribute__((ext_vector_type(16))) __bf16 bf16x16;
typedef __attribute__((ext_vector_type(8)))  float  f32x8;
typedef __attribute__((ext_vector_type(4)))  unsigned int uint32x4;
typedef __attribute__((ext_vector_type(8)))  int int32x8;
typedef __attribute__((ext_vector_type(4)))  int int32x4;

#define BB 8
#define SS 4096
#define CC 1024
#define HH 16
#define DD 64
#define KK 256

#if defined(__has_builtin)
#if __has_builtin(__builtin_amdgcn_tensor_load_to_lds) && \
    __has_builtin(__builtin_amdgcn_s_wait_tensorcnt)
#define USE_TDM 1
#endif
#endif
#ifndef USE_TDM
#define USE_TDM 0
#endif

// ---------------------------- small helper kernels -------------------------

__global__ void cvt_f32_bf16(const float* __restrict__ in, __bf16* __restrict__ out, int n) {
  int i = blockIdx.x * blockDim.x + threadIdx.x;
  if (i < n) out[i] = (__bf16)in[i];
}

// Wf[he, c] = sum_d head_w[he, d] * base_w[d, c]   (fuses to_q_w with wq_w etc.)
__global__ void fuse_head_weights(const float* __restrict__ head_w,  // [H*D, D]
                                  const float* __restrict__ base_w,  // [D, C]
                                  __bf16* __restrict__ out) {        // [H*D, C] bf16
  int idx = blockIdx.x * blockDim.x + threadIdx.x;   // over 1024*1024
  int c  = idx & (CC - 1);
  int he = idx >> 10;
  float s = 0.f;
#pragma unroll 8
  for (int d = 0; d < DD; ++d) s += head_w[he * DD + d] * base_w[d * CC + c];
  out[idx] = (__bf16)s;
}

// ------------------------------ WMMA fragments -----------------------------
// CDNA5 16-bit VGPR layouts (ISA 7.12.2):
//  A 16x32: lane m=lane&15, half=lane>>4 holds K = half*8+[0..7], 16+half*8+[0..7]
//  B 32x16: lane n=lane&15, half=lane>>4 holds K = half*8+[0..7], 16+half*8+[0..7]
// NT form (B stored [N,Kc] row-major): contiguous 16B/32B runs; works for both
// global and LDS source pointers (generic address space).

__device__ inline bf16x16 load_a_frag(const __bf16* A, int lda, int r0, int k0, int lane) {
  const int m = lane & 15, hf = lane >> 4;
  const __bf16* p = A + (size_t)(r0 + m) * lda + k0 + hf * 8;
  union { uint4 u[2]; bf16x16 v; } f;
  f.u[0] = *reinterpret_cast<const uint4*>(p);
  f.u[1] = *reinterpret_cast<const uint4*>(p + 16);
  return f.v;
}

__device__ inline bf16x16 load_b_frag(const __bf16* Bm, int ldb, int n0, int k0, int lane) {
  const int n = lane & 15, hf = lane >> 4;
  const __bf16* p = Bm + (size_t)(n0 + n) * ldb + k0 + hf * 8;
  union { uint4 u[2]; bf16x16 v; } f;
  f.u[0] = *reinterpret_cast<const uint4*>(p);
  f.u[1] = *reinterpret_cast<const uint4*>(p + 16);
  return f.v;
}

__device__ inline float gelu_erf(float v) {
  return 0.5f * v * (1.f + erff(v * 0.70710678118654752f));
}

#if USE_TDM
// TDM: DMA one contiguous 32KB row (16384 x 2B) from global into LDS at
// lds_byte_off. D# bitfields per CDNA5 ISA 08_async_tensor 8.3/8.4:
//  group0: [1:0]=count=1 | [63:32]=lds_addr | [120:64]=global_addr | [127:126]=type=2
//  group1: [17:16]=data_size=1(2B) | tensor_dim0=16384 | tensor_dim1=1 |
//          tile_dim0=16384 | tile_dim1=1 | tensor_dim0_stride=16384
// This toolchain's builtin takes 6 args (groups 0..3 + extra group + cpol).
__device__ inline void tdm_load_32kb(const __bf16* gsrc, unsigned lds_byte_off) {
  const unsigned long long ga = (unsigned long long)(size_t)gsrc;
  uint32x4 g0 = { 1u,                                  // count=1, user desc
                  lds_byte_off,                        // lds_addr
                  (unsigned)ga,                        // global_addr[31:0]
                  ((unsigned)(ga >> 32) & 0x01FFFFFFu) // global_addr[56:32]
                      | 0x80000000u };                 // type=2 ("image")
  int32x8 g1 = { 0x00010000,   // data_size=1 (2 bytes), mask=0
                 0x40000000,   // tensor_dim0[15:0]=16384 in [31:16]
                 0x00010000,   // tensor_dim1[15:0]=1 in [31:16]
                 0x40000000,   // tile_dim0=16384 in [31:16]
                 0x00000001,   // tile_dim1=1
                 0x00004000,   // tensor_dim0_stride=16384
                 0, 0 };
  int32x4 z4 = { 0, 0, 0, 0 };
  int32x8 z8 = { 0, 0, 0, 0, 0, 0, 0, 0 };
  __builtin_amdgcn_tensor_load_to_lds(g0, g1, z4, z4, z8, 0);
}
#endif

// ------------------------------ WMMA NT GEMM -------------------------------

enum { OM_F32 = 0, OM_BF16 = 1, OM_BF16T = 2, OM_QW = 3, OM_KT = 4 };
enum { BIAS_NONE = 0, BIAS_ROW = 1, BIAS_COL = 2 };

// Block: 256 threads = 8 waves; tile 64(M) x 64(N); wave (w&3)->16 rows,
// (w>>2)->32 cols as two 16x16 WMMA subtiles. Requires M%64==0, N%64==0, Kc%32==0.
template <int OM, int BMODE>
__global__ __launch_bounds__(256)
void gemm_nt_wmma(const __bf16* __restrict__ A, int lda, long long sA,
                  const __bf16* __restrict__ Bm, int ldb, long long sB,
                  void* __restrict__ Cp, int ldc, long long sC,
                  const float* __restrict__ bias, float alpha, int Kc) {
  const int z = blockIdx.z;
  A  += (long long)z * sA;
  Bm += (long long)z * sB;
  const int lane = threadIdx.x & 31;
  const int w    = threadIdx.x >> 5;
  const int rowBase = blockIdx.y * 64 + (w & 3) * 16;
  const int colBase = blockIdx.x * 64 + (w >> 2) * 32;

  f32x8 acc0 = {0.f, 0.f, 0.f, 0.f, 0.f, 0.f, 0.f, 0.f};
  f32x8 acc1 = acc0;
  for (int k0 = 0; k0 < Kc; k0 += 32) {
    bf16x16 a  = load_a_frag(A, lda, rowBase, k0, lane);
    bf16x16 b0 = load_b_frag(Bm, ldb, colBase,      k0, lane);
    bf16x16 b1 = load_b_frag(Bm, ldb, colBase + 16, k0, lane);
    acc0 = __builtin_amdgcn_wmma_f32_16x16x32_bf16(false, a, false, b0, (short)0, acc0, false, false);
    acc1 = __builtin_amdgcn_wmma_f32_16x16x32_bf16(false, a, false, b1, (short)0, acc1, false, false);
  }

  const int i = lane & 15, hf = lane >> 4;
#pragma unroll
  for (int t = 0; t < 2; ++t) {
    const f32x8 acc = t ? acc1 : acc0;
    const int n  = colBase + t * 16 + i;
    const int m0 = rowBase + hf * 8;
    float vals[8];
#pragma unroll
    for (int r = 0; r < 8; ++r) {
      float v = acc[r] * alpha;
      if constexpr (BMODE == BIAS_ROW) v += bias[m0 + r];
      if constexpr (BMODE == BIAS_COL) v += bias[n];
      vals[r] = v;
    }
    if constexpr (OM == OM_F32) {
      float* out = (float*)Cp + (long long)z * sC;
#pragma unroll
      for (int r = 0; r < 8; ++r) out[(size_t)(m0 + r) * ldc + n] = vals[r];
    } else if constexpr (OM == OM_BF16) {
      __bf16* out = (__bf16*)Cp + (long long)z * sC;
#pragma unroll
      for (int r = 0; r < 8; ++r) out[(size_t)(m0 + r) * ldc + n] = (__bf16)vals[r];
    } else if constexpr (OM == OM_BF16T) {
      // transposed write: lane's 8 rows are contiguous -> one 16B store
      __bf16* out = (__bf16*)Cp + (long long)z * sC;
      union { uint4 u; __bf16 b[8]; } pk;
#pragma unroll
      for (int r = 0; r < 8; ++r) pk.b[r] = (__bf16)vals[r];
      *reinterpret_cast<uint4*>(out + (size_t)n * ldc + m0) = pk.u;
    } else if constexpr (OM == OM_QW) {        // z=b: [B,H,S,D]
      __bf16* out = (__bf16*)Cp;
      const int h = n >> 6, e = n & 63;
#pragma unroll
      for (int r = 0; r < 8; ++r)
        out[(((size_t)z * HH + h) * SS + (m0 + r)) * DD + e] = (__bf16)vals[r];
    } else {                                   // OM_KT: z=b: [B,H,D,S], packed
      __bf16* out = (__bf16*)Cp;
      const int h = n >> 6, e = n & 63;
      union { uint4 u; __bf16 b[8]; } pk;
#pragma unroll
      for (int r = 0; r < 8; ++r) pk.b[r] = (__bf16)vals[r];
      *reinterpret_cast<uint4*>(out + (((size_t)z * HH + h) * DD + e) * SS + m0) = pk.u;
    }
  }
}

// --------------------- fused attention (scores+softmax+PV) -----------------
// Grid: x = S/64 (row tile), y = b*H+h. Block 256 = 8 waves.
// LDS: sKE [K=256][D=64] (32KB) | sVF [D=64][K=256] (32KB) | sP [64][256] (32KB)
// Phase 0: stage KEt/VFt via TDM (wave 0, s_wait_tensorcnt) or coop copy
// Phase 1: scores 64x256 = QW_tile x sKE^T via WMMA (8 waves x 8 acc), /sqrt(D)
// Phase 2: softmax over K in LDS (wave per 8 rows)
// Phase 3: out 64x64 = sP x sVF^T via WMMA, GELU, scatter to [B,S,H*D]

__global__ __launch_bounds__(256)
void attn_fused(const __bf16* __restrict__ QW,   // [B,H,S,D]
                const __bf16* __restrict__ KEt,  // [B,H,K,D]
                const __bf16* __restrict__ VFt,  // [B,H,D,K]
                __bf16* __restrict__ Gout) {     // [B,S,H*D]
  extern __shared__ char smem_raw[];
  __bf16* sKE = (__bf16*)smem_raw;       // 256*64, LDS offset 0
  __bf16* sVF = sKE + KK * DD;           // 64*256, LDS offset 32768
  __bf16* sP  = sVF + DD * KK;           // 64*256, LDS offset 65536

  const int zz   = blockIdx.y;           // b*H + h
  const int tid  = threadIdx.x;
  const int lane = tid & 31;
  const int w    = tid >> 5;

  // ---- phase 0: stage KEt / VFt tiles into LDS ----
#if USE_TDM
  if (w == 0) {
    tdm_load_32kb(KEt + (size_t)zz * KK * DD, 0u);
    tdm_load_32kb(VFt + (size_t)zz * DD * KK, (unsigned)(KK * DD * 2));
    __builtin_amdgcn_s_wait_tensorcnt(0);
  }
#else
  {
    const uint4* gK = reinterpret_cast<const uint4*>(KEt + (size_t)zz * KK * DD);
    const uint4* gV = reinterpret_cast<const uint4*>(VFt + (size_t)zz * DD * KK);
    uint4* lK = reinterpret_cast<uint4*>(sKE);
    uint4* lV = reinterpret_cast<uint4*>(sVF);
#pragma unroll
    for (int j = 0; j < (KK * DD / 8) / 256; ++j) {
      lK[tid + j * 256] = gK[tid + j * 256];
      lV[tid + j * 256] = gV[tid + j * 256];
    }
  }
#endif
  __syncthreads();

  const int i = lane & 15, hf = lane >> 4;
  const int rowBase = (w & 3) * 16;       // 4 row groups of 16

  // ---- phase 1: scores tile 64x256, Kc = D = 64 ----
  {
    const __bf16* Aq = QW + ((size_t)zz * SS + (size_t)blockIdx.x * 64) * DD;
    const int colBase = (w >> 2) * 128;   // 2 col groups of 128 (8 subtiles)
    f32x8 acc[8];
#pragma unroll
    for (int j = 0; j < 8; ++j) acc[j] = f32x8{0.f,0.f,0.f,0.f,0.f,0.f,0.f,0.f};
#pragma unroll
    for (int k0 = 0; k0 < DD; k0 += 32) {
      bf16x16 a = load_a_frag(Aq, DD, rowBase, k0, lane);
#pragma unroll
      for (int j = 0; j < 8; ++j) {
        bf16x16 b = load_b_frag(sKE, DD, colBase + j * 16, k0, lane);
        acc[j] = __builtin_amdgcn_wmma_f32_16x16x32_bf16(false, a, false, b, (short)0, acc[j], false, false);
      }
    }
#pragma unroll
    for (int j = 0; j < 8; ++j)
#pragma unroll
      for (int r = 0; r < 8; ++r)
        sP[(size_t)(rowBase + hf * 8 + r) * KK + colBase + j * 16 + i] =
            (__bf16)(acc[j][r] * 0.125f);   // 1/sqrt(D)
  }
  __syncthreads();

  // ---- phase 2: softmax over K=256, wave handles rows [w*8, w*8+8) ----
#pragma unroll
  for (int rr = 0; rr < 8; ++rr) {
    __bf16* p = sP + (size_t)(w * 8 + rr) * KK + lane * 8;
    union { uint4 u; __bf16 b[8]; } ld;
    ld.u = *reinterpret_cast<const uint4*>(p);
    float v[8], mx = -3.0e38f;
#pragma unroll
    for (int j = 0; j < 8; ++j) { v[j] = (float)ld.b[j]; mx = fmaxf(mx, v[j]); }
#pragma unroll
    for (int off = 16; off > 0; off >>= 1) mx = fmaxf(mx, __shfl_xor(mx, off, 32));
    float sum = 0.f;
#pragma unroll
    for (int j = 0; j < 8; ++j) { v[j] = __expf(v[j] - mx); sum += v[j]; }
#pragma unroll
    for (int off = 16; off > 0; off >>= 1) sum += __shfl_xor(sum, off, 32);
    const float inv = 1.f / sum;
    union { uint4 u; __bf16 b[8]; } st;
#pragma unroll
    for (int j = 0; j < 8; ++j) st.b[j] = (__bf16)(v[j] * inv);
    *reinterpret_cast<uint4*>(p) = st.u;
  }
  __syncthreads();

  // ---- phase 3: out tile 64x64 = P[64,256] x VF[64,256]^T, GELU, scatter ----
  {
    const int colBase = (w >> 2) * 32;
    f32x8 o0 = {0.f,0.f,0.f,0.f,0.f,0.f,0.f,0.f};
    f32x8 o1 = o0;
#pragma unroll
    for (int k0 = 0; k0 < KK; k0 += 32) {
      bf16x16 a  = load_a_frag(sP, KK, rowBase, k0, lane);
      bf16x16 b0 = load_b_frag(sVF, KK, colBase,      k0, lane);
      bf16x16 b1 = load_b_frag(sVF, KK, colBase + 16, k0, lane);
      o0 = __builtin_amdgcn_wmma_f32_16x16x32_bf16(false, a, false, b0, (short)0, o0, false, false);
      o1 = __builtin_amdgcn_wmma_f32_16x16x32_bf16(false, a, false, b1, (short)0, o1, false, false);
    }
    const int bb = zz >> 4, h = zz & 15;
#pragma unroll
    for (int t = 0; t < 2; ++t) {
      const f32x8 o = t ? o1 : o0;
      const int n = colBase + t * 16 + i;
#pragma unroll
      for (int r = 0; r < 8; ++r) {
        const size_t m = (size_t)blockIdx.x * 64 + rowBase + hf * 8 + r;
        Gout[((size_t)bb * SS + m) * (HH * DD) + h * DD + n] = (__bf16)gelu_erf(o[r]);
      }
    }
  }
}

// ------------------------------- launcher ----------------------------------

extern "C" void kernel_launch(void* const* d_in, const int* in_sizes, int n_in,
                              void* d_out, int out_size, void* d_ws, size_t ws_size,
                              hipStream_t stream) {
  (void)in_sizes; (void)n_in; (void)out_size; (void)ws_size;
  const float* tensor = (const float*)d_in[0];
  const float* to_q_w = (const float*)d_in[1];
  const float* to_k_w = (const float*)d_in[2];
  const float* to_v_w = (const float*)d_in[3];
  const float* wq_w   = (const float*)d_in[4];
  const float* wq_b   = (const float*)d_in[5];
  const float* wk_w   = (const float*)d_in[6];
  const float* wk_b   = (const float*)d_in[7];
  const float* wv_w   = (const float*)d_in[8];
  const float* wv_b   = (const float*)d_in[9];
  const float* E_w    = (const float*)d_in[10];
  const float* E_b    = (const float*)d_in[11];
  const float* F_w    = (const float*)d_in[12];
  const float* F_b    = (const float*)d_in[13];
  const float* wo_w   = (const float*)d_in[14];
  const float* wo_b   = (const float*)d_in[15];
  float* out = (float*)d_out;

  // ---- workspace bump allocator (256B aligned), total ~350 MB ----
  char* base = (char*)d_ws;
  size_t off = 0;
  auto alloc = [&](size_t bytes) -> void* {
    void* p = base + off;
    off += (bytes + 255) & ~(size_t)255;
    return p;
  };
  __bf16* tbf  = (__bf16*)alloc((size_t)BB * SS * CC * 2);       // tensor bf16
  __bf16* Wqf  = (__bf16*)alloc((size_t)HH * DD * CC * 2);       // fused Q weights
  __bf16* Wkf  = (__bf16*)alloc((size_t)HH * DD * CC * 2);
  __bf16* Wvf  = (__bf16*)alloc((size_t)HH * DD * CC * 2);
  __bf16* Ebf  = (__bf16*)alloc((size_t)KK * SS * 2);
  __bf16* Fbf  = (__bf16*)alloc((size_t)KK * SS * 2);
  __bf16* Wobf = (__bf16*)alloc((size_t)CC * HH * DD * 2);
  __bf16* QW   = (__bf16*)alloc((size_t)BB * HH * SS * DD * 2);  // [B,H,S,D]
  __bf16* KWt  = (__bf16*)alloc((size_t)BB * HH * DD * SS * 2);  // [B,H,D,S]
  __bf16* VWt  = (__bf16*)alloc((size_t)BB * HH * DD * SS * 2);  // [B,H,D,S]
  __bf16* KEt  = (__bf16*)alloc((size_t)BB * HH * KK * DD * 2);  // [B,H,K,D]
  __bf16* VFt  = (__bf16*)alloc((size_t)BB * HH * DD * KK * 2);  // [B,H,D,K]
  __bf16* Gout = (__bf16*)alloc((size_t)BB * SS * HH * DD * 2);  // gelu(concat)

  // ---- precompute: conversions + fused projection weights ----
  {
    int n = BB * SS * CC;
    cvt_f32_bf16<<<(n + 255) / 256, 256, 0, stream>>>(tensor, tbf, n);
    n = KK * SS;
    cvt_f32_bf16<<<(n + 255) / 256, 256, 0, stream>>>(E_w, Ebf, n);
    cvt_f32_bf16<<<(n + 255) / 256, 256, 0, stream>>>(F_w, Fbf, n);
    n = CC * HH * DD;
    cvt_f32_bf16<<<(n + 255) / 256, 256, 0, stream>>>(wo_w, Wobf, n);
    fuse_head_weights<<<(HH * DD * CC) / 256, 256, 0, stream>>>(wq_w, to_q_w, Wqf);
    fuse_head_weights<<<(HH * DD * CC) / 256, 256, 0, stream>>>(wk_w, to_k_w, Wkf);
    fuse_head_weights<<<(HH * DD * CC) / 256, 256, 0, stream>>>(wv_w, to_v_w, Wvf);
  }

  // ---- projections: per batch b, [S x C] x [H*D x C]^T -> heads layouts ----
  {
    dim3 grid(CC / 64, SS / 64, BB);
    gemm_nt_wmma<OM_QW, BIAS_COL><<<grid, 256, 0, stream>>>(
        tbf, CC, (long long)SS * CC, Wqf, CC, 0, QW, 0, 0, wq_b, 1.f, CC);
    gemm_nt_wmma<OM_KT, BIAS_COL><<<grid, 256, 0, stream>>>(
        tbf, CC, (long long)SS * CC, Wkf, CC, 0, KWt, 0, 0, wk_b, 1.f, CC);
    gemm_nt_wmma<OM_KT, BIAS_COL><<<grid, 256, 0, stream>>>(
        tbf, CC, (long long)SS * CC, Wvf, CC, 0, VWt, 0, 0, wv_b, 1.f, CC);
  }

  // ---- sequence compression: per (b,h), reduce over S=4096 ----
  {
    dim3 grid(DD / 64, KK / 64, BB * HH);
    // KEt[k,d] = sum_s E[k,s]*KW[s,d] + E_b[k]   (bias over rows m=k)
    gemm_nt_wmma<OM_BF16, BIAS_ROW><<<grid, 256, 0, stream>>>(
        Ebf, SS, 0, KWt, SS, (long long)DD * SS,
        KEt, DD, (long long)KK * DD, E_b, 1.f, SS);
    // VFt[d,k] = (sum_s F[k,s]*VW[s,d] + F_b[k]) written transposed (packed 16B)
    gemm_nt_wmma<OM_BF16T, BIAS_ROW><<<grid, 256, 0, stream>>>(
        Fbf, SS, 0, VWt, SS, (long long)DD * SS,
        VFt, KK, (long long)DD * KK, F_b, 1.f, SS);
  }

  // ---- fused attention: scores + softmax + P*V + GELU + concat scatter ----
  {
    dim3 grid(SS / 64, BB * HH);
    const size_t smem = (size_t)(KK * DD + DD * KK + 64 * KK) * sizeof(__bf16); // 96 KB
    attn_fused<<<grid, 256, smem, stream>>>(QW, KEt, VFt, Gout);
  }

  // ---- output projection: [B*S x H*D] x [C x H*D]^T + wo_b -> f32 out ----
  {
    dim3 grid(CC / 64, (BB * SS) / 64, 1);
    gemm_nt_wmma<OM_F32, BIAS_COL><<<grid, 256, 0, stream>>>(
        Gout, HH * DD, 0, Wobf, HH * DD, 0,
        out, CC, 0, wo_b, 1.f, HH * DD);
  }
}
